// RPNLearner_31714038514190
// MI455X (gfx1250) — compile-verified, matrix operand order checked by reading, not run on hardware
//
#include <hip/hip_runtime.h>
#include <hip/hip_bf16.h>
#include <math.h>

// ---------------------------------------------------------------------------
// Types for CDNA5 WMMA (wave32)
// ---------------------------------------------------------------------------
typedef __bf16 bf16;
typedef __attribute__((ext_vector_type(16))) __bf16 v16bf;
typedef __attribute__((ext_vector_type(8)))  __bf16 v8bf;
typedef __attribute__((ext_vector_type(8)))  float  v8f;

#define BM 128
#define BN 128
#define BK 32
#define LDSS 40   // LDS row stride in halfs (80 bytes -> 16B aligned, bank-friendly)

// CDNA5 async global->LDS copy (ASYNCcnt-tracked). Guarded so the file still
// compiles (with synchronous staging) if the toolchain lacks the builtins.
#if defined(__has_builtin)
#if __has_builtin(__builtin_amdgcn_global_load_async_to_lds_b128) && \
    __has_builtin(__builtin_amdgcn_s_wait_asynccnt)
#define USE_ASYNC_LDS 1
#endif
#endif
#ifndef USE_ASYNC_LDS
#define USE_ASYNC_LDS 0
#endif

// Builtin expects pointers to int __attribute__((vector_size(16))) in the
// global (AS1) / LDS (AS3) address spaces (per hipcc diagnostic).
typedef int v4i_gnu __attribute__((vector_size(16)));
typedef __attribute__((address_space(1))) v4i_gnu as1_v4i;
typedef __attribute__((address_space(3))) v4i_gnu as3_v4i;

__device__ __forceinline__ void stage_b128(const bf16* gsrc, bf16* ldst) {
#if USE_ASYNC_LDS
  __builtin_amdgcn_global_load_async_to_lds_b128(
      (as1_v4i*)gsrc, (as3_v4i*)ldst, 0, 0);
#else
  *(uint4*)ldst = *(const uint4*)gsrc;
#endif
}

enum {
  EPI_BN_RELU = 0,     // (acc + conv_b - mean) * gamma*rsqrt(var+eps) + beta, relu -> bf16
  EPI_BIAS,            // acc + bias -> bf16
  EPI_SCALE_F32,       // acc * alpha -> f32
  EPI_DUAL_PLAIN,      // acc -> f32 and bf16
  EPI_BIAS_RELU,       // relu(acc + bias) -> bf16
  EPI_BIAS_DUAL,       // acc + bias -> f32 and bf16
  EPI_BIAS_TANH_F32    // tanh(acc + bias) -> f32
};

// Load a 16x32 bf16 WMMA A/B fragment row from LDS (ISA 7.12.2 layout):
// lanes 0-15 hold K in [0,8)+[16,24), lanes 16-31 hold K in [8,16)+[24,32).
__device__ __forceinline__ v16bf load_frag(const bf16* rowp, int khalf) {
  v8bf lo = *(const v8bf*)(rowp + khalf * 8);
  v8bf hi = *(const v8bf*)(rowp + 16 + khalf * 8);
  v16bf r;
#pragma unroll
  for (int e = 0; e < 8; ++e) { r[e] = lo[e]; r[8 + e] = hi[e]; }
  return r;
}

// ---------------------------------------------------------------------------
// Tiled bf16 GEMM: C[M,N] = A[M,K] * B[K,N], with B ALWAYS stored [N][K]
// row-major (ldb = row stride of B^T). Block tile 128x128x32, 256 threads =
// 8 waves (wave32), each wave computes 2x4 WMMA 16x16 tiles. Double-buffered
// LDS with async global->LDS staging (prefetch next K-tile during WMMA).
// Batched via rows_per_batch + element strides; block rows never straddle a
// batch (rows_per_batch is a multiple of BM everywhere here).
// TRANS_OUT: store bf16 result batch-transposed: out[b][n][row_in_batch].
// ---------------------------------------------------------------------------
template <int EPI, bool TRANS_OUT>
__global__ __launch_bounds__(256) void gemm_bf16(
    const bf16* __restrict__ A, const bf16* __restrict__ Bt,
    float* __restrict__ outF, bf16* __restrict__ outB,
    const float* __restrict__ p0, const float* __restrict__ p1,
    const float* __restrict__ p2, const float* __restrict__ p3,
    const float* __restrict__ p4,
    int N, int K, int lda, int ldb, int ldc,
    int rows_per_batch, long long a_bstride, long long b_bstride,
    float alpha)
{
  __shared__ __align__(16) bf16 sA[2][BM * LDSS];
  __shared__ __align__(16) bf16 sB[2][BN * LDSS];

  const int tid    = threadIdx.x;
  const int lane   = tid & 31;
  const int wave   = tid >> 5;
  const int wm     = wave >> 1;   // 0..3 : 32-row strip
  const int wn     = wave & 1;    // 0..1 : 64-col strip
  const int lane16 = lane & 15;
  const int khalf  = lane >> 4;

  const int row0  = blockIdx.y * BM;
  const int col0  = blockIdx.x * BN;
  const int batch = row0 / rows_per_batch;
  const int lrow0 = row0 - batch * rows_per_batch;

  const bf16* Ab = A  + (long long)batch * a_bstride + (long long)lrow0 * lda;
  const bf16* Bb = Bt + (long long)batch * b_bstride;

  // Per-thread staging coordinates: 2 x b128 for A, 2 x b128 for B per K-tile.
  const int sr = tid >> 2;        // 0..63
  const int sc = (tid & 3) * 8;   // 0,8,16,24

  auto issue = [&](int k0, int buf) {
#pragma unroll
    for (int p = 0; p < 2; ++p) {
      const int r = sr + p * 64;
      stage_b128(Ab + (long long)r * lda + (k0 + sc), &sA[buf][r * LDSS + sc]);
      stage_b128(Bb + (long long)(col0 + r) * ldb + (k0 + sc),
                 &sB[buf][r * LDSS + sc]);
    }
  };

  v8f acc[2][4];
#pragma unroll
  for (int im = 0; im < 2; ++im)
#pragma unroll
    for (int in = 0; in < 4; ++in)
#pragma unroll
      for (int e = 0; e < 8; ++e) acc[im][in][e] = 0.0f;

  const int nk = K / BK;
  issue(0, 0);
  for (int i = 0; i < nk; ++i) {
    const int buf = i & 1;
    if (i + 1 < nk) {
      issue((i + 1) * BK, buf ^ 1);   // prefetch next tile into other buffer
#if USE_ASYNC_LDS
      __builtin_amdgcn_s_wait_asynccnt(4);  // current tile's 4 copies done
#endif
    } else {
#if USE_ASYNC_LDS
      __builtin_amdgcn_s_wait_asynccnt(0);
#endif
    }
    __syncthreads();   // all waves' tile-i data visible in LDS

    v16bf af[2], bfr[4];
#pragma unroll
    for (int im = 0; im < 2; ++im)
      af[im] = load_frag(&sA[buf][(wm * 32 + im * 16 + lane16) * LDSS], khalf);
#pragma unroll
    for (int in = 0; in < 4; ++in)
      bfr[in] = load_frag(&sB[buf][(wn * 64 + in * 16 + lane16) * LDSS], khalf);

#pragma unroll
    for (int im = 0; im < 2; ++im)
#pragma unroll
      for (int in = 0; in < 4; ++in)
        acc[im][in] = __builtin_amdgcn_wmma_f32_16x16x32_bf16(
            false, af[im], false, bfr[in], (short)0, acc[im][in], false, false);

    __syncthreads();   // done reading buf -> safe to overwrite next iteration
  }

  // ---- epilogue (C/D layout: lane n = lane&15, VGPR j -> row j + 8*(lane>=16)) ----
#pragma unroll
  for (int im = 0; im < 2; ++im) {
#pragma unroll
    for (int in = 0; in < 4; ++in) {
      const int col = col0 + wn * 64 + in * 16 + lane16;
      float b0 = 0.f, scl = 0.f, bt = 0.f, mn = 0.f;
      if constexpr (EPI == EPI_BN_RELU) {
        b0  = p0[col];
        scl = p1[col] * rsqrtf(p4[col] + 1e-5f);
        bt  = p2[col];
        mn  = p3[col];
      } else if constexpr (EPI == EPI_BIAS || EPI == EPI_BIAS_RELU ||
                           EPI == EPI_BIAS_DUAL || EPI == EPI_BIAS_TANH_F32) {
        b0 = p0[col];
      }
#pragma unroll
      for (int j = 0; j < 8; ++j) {
        const int lr  = lrow0 + wm * 32 + im * 16 + khalf * 8 + j;
        const int row = row0 + wm * 32 + im * 16 + khalf * 8 + j;
        const long long o = (long long)row * ldc + col;
        // batch-transposed bf16 store target: out[b][col][lr]
        const long long ot = (long long)batch * ((long long)N * rows_per_batch)
                           + (long long)col * rows_per_batch + lr;
        float v = acc[im][in][j];
        if constexpr (EPI == EPI_BN_RELU) {
          v = (v + b0 - mn) * scl + bt;
          v = fmaxf(v, 0.f);
          outB[TRANS_OUT ? ot : o] = (bf16)v;
        } else if constexpr (EPI == EPI_BIAS) {
          outB[TRANS_OUT ? ot : o] = (bf16)(v + b0);
        } else if constexpr (EPI == EPI_SCALE_F32) {
          outF[o] = v * alpha;
        } else if constexpr (EPI == EPI_DUAL_PLAIN) {
          outF[o] = v;
          outB[o] = (bf16)v;
        } else if constexpr (EPI == EPI_BIAS_RELU) {
          v = fmaxf(v + b0, 0.f);
          outB[TRANS_OUT ? ot : o] = (bf16)v;
        } else if constexpr (EPI == EPI_BIAS_DUAL) {
          v += b0;
          outF[o] = v;
          outB[o] = (bf16)v;
        } else { // EPI_BIAS_TANH_F32
          outF[o] = tanhf(v + b0);
        }
      }
    }
  }
}

// ---------------------------------------------------------------------------
// [C, S] (xB batches) f32 -> [S, C] bf16 transpose-convert via 32x32 LDS tile.
// Used both for the input reshape ([B,C,S] -> [B,S,C]) and to pre-transpose
// every weight matrix ([K,N] -> [N,K]) so GEMM B-staging is a straight copy.
// ---------------------------------------------------------------------------
__global__ __launch_bounds__(256) void k_transpose_bf16(
    const float* __restrict__ in, bf16* __restrict__ out, int C, int S) {
  __shared__ float tile[32][33];
  const int b  = blockIdx.z;
  const int s0 = blockIdx.x * 32;
  const int c0 = blockIdx.y * 32;
  const float* ip = in + (long long)b * C * S;
  bf16* op = out + (long long)b * S * C;
#pragma unroll
  for (int r = 0; r < 32; r += 8)
    tile[threadIdx.y + r][threadIdx.x] =
        ip[(long long)(c0 + threadIdx.y + r) * S + s0 + threadIdx.x];
  __syncthreads();
#pragma unroll
  for (int r = 0; r < 32; r += 8)
    op[(long long)(s0 + threadIdx.y + r) * C + c0 + threadIdx.x] =
        (bf16)tile[threadIdx.x][threadIdx.y + r];
}

// ---------------------------------------------------------------------------
// Row softmax over 1024 f32 scores -> bf16 attn. One 256-thread block per row.
// ---------------------------------------------------------------------------
__global__ __launch_bounds__(256) void k_softmax_row(
    const float* __restrict__ s, bf16* __restrict__ a, int T) {
  __shared__ float red[256];
  const long long row = blockIdx.x;
  const float* sr = s + row * T;
  bf16* ar = a + row * T;
  float x[4];
  float mx = -3.4e38f;
#pragma unroll
  for (int i = 0; i < 4; ++i) {
    x[i] = sr[threadIdx.x + i * 256];
    mx = fmaxf(mx, x[i]);
  }
  red[threadIdx.x] = mx;
  __syncthreads();
  for (int off = 128; off > 0; off >>= 1) {
    if (threadIdx.x < off)
      red[threadIdx.x] = fmaxf(red[threadIdx.x], red[threadIdx.x + off]);
    __syncthreads();
  }
  mx = red[0];
  __syncthreads();
  float sm = 0.f;
#pragma unroll
  for (int i = 0; i < 4; ++i) {
    x[i] = __expf(x[i] - mx);
    sm += x[i];
  }
  red[threadIdx.x] = sm;
  __syncthreads();
  for (int off = 128; off > 0; off >>= 1) {
    if (threadIdx.x < off) red[threadIdx.x] += red[threadIdx.x + off];
    __syncthreads();
  }
  const float inv = 1.0f / red[0];
#pragma unroll
  for (int i = 0; i < 4; ++i)
    ar[threadIdx.x + i * 256] = (bf16)(x[i] * inv);
}

// ---------------------------------------------------------------------------
// Host-side orchestration
// ---------------------------------------------------------------------------
extern "C" void kernel_launch(void* const* d_in, const int* in_sizes, int n_in,
                              void* d_out, int out_size, void* d_ws, size_t ws_size,
                              hipStream_t stream) {
  (void)in_sizes; (void)n_in; (void)out_size; (void)ws_size;

  const float* x      = (const float*)d_in[0];
  const float* y      = (const float*)d_in[1];
  const float* conv_w = (const float*)d_in[2];
  const float* conv_b = (const float*)d_in[3];
  const float* gamma  = (const float*)d_in[4];
  const float* beta   = (const float*)d_in[5];
  const float* mean   = (const float*)d_in[6];
  const float* var    = (const float*)d_in[7];
  const float* wq     = (const float*)d_in[8];
  const float* bq     = (const float*)d_in[9];
  const float* wk     = (const float*)d_in[10];
  const float* bk     = (const float*)d_in[11];
  const float* wv     = (const float*)d_in[12];
  const float* bv     = (const float*)d_in[13];
  const float* enc_w  = (const float*)d_in[14];
  const float* enc_b  = (const float*)d_in[15];
  const float* mu_w   = (const float*)d_in[16];
  const float* mu_b   = (const float*)d_in[17];
  const float* dec_w  = (const float*)d_in[18];
  const float* dec_b  = (const float*)d_in[19];
  const float* out_w  = (const float*)d_in[20];
  const float* out_b  = (const float*)d_in[21];

  float* out = (float*)d_out;
  char*  ws  = (char*)d_ws;

  const int Bc = 16, CIN = 256, S = 1024, HID = 512, LAT = 256;
  const int M = Bc * S; // 16384

  // Workspace layout (bytes). Overlapped: scores reuses dead xrm/yrm/xp/yp;
  // attn reuses dead q/k; h/z/g reuse dead scores. Total ~137.6 MB.
  const size_t OFF_XRM   = 0;              // 16384*256  bf16
  const size_t OFF_YRM   = 8388608;
  const size_t OFF_XP    = 16777216;       // 16384*512  bf16
  const size_t OFF_YP    = 33554432;
  const size_t OFF_SCORES= 0;              // 16*1024*1024 f32 overlays above
  const size_t OFF_Q     = 67108864;       // 16384*512  bf16
  const size_t OFF_K     = 83886080;
  const size_t OFF_VT    = 100663296;      // V^T: [16][512][1024] bf16
  const size_t OFF_ATTN  = 67108864;       // 16*1024*1024 bf16 overlays Q,K
  const size_t OFF_YPROJ = 117440512;      // 16384*512  bf16
  const size_t OFF_H     = 0;              // overlays dead scores
  const size_t OFF_ZBF   = 16777216;
  const size_t OFF_G     = 25165824;
  const size_t OFF_WCONV = 134217728;      // all weights stored TRANSPOSED [N][K]
  const size_t OFF_WQ    = 134479872;
  const size_t OFF_WK    = 135004160;
  const size_t OFF_WV    = 135528448;
  const size_t OFF_WENC  = 136052736;
  const size_t OFF_WMU   = 136577024;
  const size_t OFF_WDEC  = 136839168;
  const size_t OFF_WOUT  = 137101312;

  // d_out: y_pred [16,1024,512] | y_projected [16,1024,512] | z [16,1024,256]
  float* out_ypred = out;
  float* out_yproj = out + 8388608;
  float* out_z     = out + 16777216;

  // ---- weights: f32 [K,N] -> bf16 [N,K] (transpose-convert) ----
  auto wtrans = [&](const float* src, size_t off, int Kd, int Nd) {
    k_transpose_bf16<<<dim3(Nd / 32, Kd / 32, 1), dim3(32, 8), 0, stream>>>(
        src, (bf16*)(ws + off), Kd, Nd);
  };
  wtrans(conv_w, OFF_WCONV, CIN, HID);
  wtrans(wq,     OFF_WQ,    HID, HID);
  wtrans(wk,     OFF_WK,    HID, HID);
  wtrans(wv,     OFF_WV,    HID, HID);
  wtrans(enc_w,  OFF_WENC,  HID, HID);
  wtrans(mu_w,   OFF_WMU,   HID, LAT);
  wtrans(dec_w,  OFF_WDEC,  LAT, HID);
  wtrans(out_w,  OFF_WOUT,  HID, HID);

  // ---- inputs: [B,C,S] f32 -> [B,S,C] bf16 ----
  {
    dim3 tb(32, 8);
    dim3 tg(S / 32, CIN / 32, Bc);
    k_transpose_bf16<<<tg, tb, 0, stream>>>(x, (bf16*)(ws + OFF_XRM), CIN, S);
    k_transpose_bf16<<<tg, tb, 0, stream>>>(y, (bf16*)(ws + OFF_YRM), CIN, S);
  }

  const dim3 blk(256);
  // G1/G2: projector (1x1 conv + BN + ReLU)
  gemm_bf16<EPI_BN_RELU, false><<<dim3(HID / BN, M / BM), blk, 0, stream>>>(
      (const bf16*)(ws + OFF_XRM), (const bf16*)(ws + OFF_WCONV),
      nullptr, (bf16*)(ws + OFF_XP),
      conv_b, gamma, beta, mean, var,
      HID, CIN, CIN, CIN, HID, M, 0LL, 0LL, 0.f);
  gemm_bf16<EPI_BN_RELU, false><<<dim3(HID / BN, M / BM), blk, 0, stream>>>(
      (const bf16*)(ws + OFF_YRM), (const bf16*)(ws + OFF_WCONV),
      nullptr, (bf16*)(ws + OFF_YP),
      conv_b, gamma, beta, mean, var,
      HID, CIN, CIN, CIN, HID, M, 0LL, 0LL, 0.f);

  // G3/G4: Q, K projections
  gemm_bf16<EPI_BIAS, false><<<dim3(HID / BN, M / BM), blk, 0, stream>>>(
      (const bf16*)(ws + OFF_XP), (const bf16*)(ws + OFF_WQ),
      nullptr, (bf16*)(ws + OFF_Q),
      bq, nullptr, nullptr, nullptr, nullptr,
      HID, HID, HID, HID, HID, M, 0LL, 0LL, 0.f);
  gemm_bf16<EPI_BIAS, false><<<dim3(HID / BN, M / BM), blk, 0, stream>>>(
      (const bf16*)(ws + OFF_YP), (const bf16*)(ws + OFF_WK),
      nullptr, (bf16*)(ws + OFF_K),
      bk, nullptr, nullptr, nullptr, nullptr,
      HID, HID, HID, HID, HID, M, 0LL, 0LL, 0.f);

  // G5: V projection, stored batch-TRANSPOSED -> V^T [b][n][s]
  gemm_bf16<EPI_BIAS, true><<<dim3(HID / BN, M / BM), blk, 0, stream>>>(
      (const bf16*)(ws + OFF_YP), (const bf16*)(ws + OFF_WV),
      nullptr, (bf16*)(ws + OFF_VT),
      bv, nullptr, nullptr, nullptr, nullptr,
      HID, HID, HID, HID, HID, S, (long long)S * HID, 0LL, 0.f);

  // G6: scores = Q K^T / sqrt(HID), batched (K already [N,K] layout)
  gemm_bf16<EPI_SCALE_F32, false><<<dim3(S / BN, M / BM), blk, 0, stream>>>(
      (const bf16*)(ws + OFF_Q), (const bf16*)(ws + OFF_K),
      (float*)(ws + OFF_SCORES), nullptr,
      nullptr, nullptr, nullptr, nullptr, nullptr,
      S, HID, HID, HID, S, S, (long long)S * HID, (long long)S * HID,
      0.04419417382415922f /* 1/sqrt(512) */);

  // softmax rows -> bf16 attn
  k_softmax_row<<<M, 256, 0, stream>>>((const float*)(ws + OFF_SCORES),
                                       (bf16*)(ws + OFF_ATTN), S);

  // G7: y_projected = attn * V, batched; B = V^T [512][1024] per batch
  gemm_bf16<EPI_DUAL_PLAIN, false><<<dim3(HID / BN, M / BM), blk, 0, stream>>>(
      (const bf16*)(ws + OFF_ATTN), (const bf16*)(ws + OFF_VT),
      out_yproj, (bf16*)(ws + OFF_YPROJ),
      nullptr, nullptr, nullptr, nullptr, nullptr,
      HID, S, S, S, HID, S, (long long)S * S, (long long)HID * S, 0.f);

  // G8: h = relu(y_projected * enc_w + enc_b)
  gemm_bf16<EPI_BIAS_RELU, false><<<dim3(HID / BN, M / BM), blk, 0, stream>>>(
      (const bf16*)(ws + OFF_YPROJ), (const bf16*)(ws + OFF_WENC),
      nullptr, (bf16*)(ws + OFF_H),
      enc_b, nullptr, nullptr, nullptr, nullptr,
      HID, HID, HID, HID, HID, M, 0LL, 0LL, 0.f);

  // G9: z = h * mu_w + mu_b (dual store f32 + bf16)
  gemm_bf16<EPI_BIAS_DUAL, false><<<dim3(LAT / BN, M / BM), blk, 0, stream>>>(
      (const bf16*)(ws + OFF_H), (const bf16*)(ws + OFF_WMU),
      out_z, (bf16*)(ws + OFF_ZBF),
      mu_b, nullptr, nullptr, nullptr, nullptr,
      LAT, HID, HID, HID, LAT, M, 0LL, 0LL, 0.f);

  // G10: g = relu(z * dec_w + dec_b)
  gemm_bf16<EPI_BIAS_RELU, false><<<dim3(HID / BN, M / BM), blk, 0, stream>>>(
      (const bf16*)(ws + OFF_ZBF), (const bf16*)(ws + OFF_WDEC),
      nullptr, (bf16*)(ws + OFF_G),
      dec_b, nullptr, nullptr, nullptr, nullptr,
      HID, LAT, LAT, LAT, HID, M, 0LL, 0LL, 0.f);

  // G11: y_pred = tanh(g * out_w + out_b)
  gemm_bf16<EPI_BIAS_TANH_F32, false><<<dim3(HID / BN, M / BM), blk, 0, stream>>>(
      (const bf16*)(ws + OFF_G), (const bf16*)(ws + OFF_WOUT),
      out_ypred, nullptr,
      out_b, nullptr, nullptr, nullptr, nullptr,
      HID, HID, HID, HID, HID, M, 0LL, 0LL, 0.f);
}
